// AttentionMessage_80135499809236
// MI455X (gfx1250) — matrix operand order repaired and checked by reading, not compiled, not run on hardware
//
#include <hip/hip_runtime.h>
#include <math.h>

#define E_EDGES 1600000
#define N_NODES 50000
#define IN_DIM  80
#define KPAD    96
#define OUT_DIM 64
#define HEADS   4

typedef __attribute__((ext_vector_type(16))) __bf16 v16bf;
typedef __attribute__((ext_vector_type(8)))  float  v8f;

union Frag {
    v16bf v;
    uint4 u[2];
};

struct Acc4 { v8f a[4]; };

// ---------- helpers ----------
__device__ __forceinline__ unsigned short f2bf(float f) {
    unsigned int u = __float_as_uint(f);
    // round-to-nearest-even
    unsigned int r = (u + 0x7FFFu + ((u >> 16) & 1u)) >> 16;
    return (unsigned short)r;
}
__device__ __forceinline__ float bf2f(unsigned short s) {
    return __uint_as_float(((unsigned int)s) << 16);
}
__device__ __forceinline__ void atomicMaxF(float* addr, float v) {
    if (v >= 0.0f) atomicMax((int*)addr, __float_as_int(v));
    else           atomicMin((unsigned int*)addr, __float_as_uint(v));
}

// Weight fragment buffer layout (ushorts):
//   wK1f[6144] wKsf[6144] wK2f[4096] wV1f[6144] wVsf[6144] wV2f[4096]
#define OFF_K1 0
#define OFF_KS 6144
#define OFF_K2 12288
#define OFF_V1 16384
#define OFF_VS 22528
#define OFF_V2 28672
#define WBUF_USHORTS 32768

#define XSTRIDE 104   // bf16 elements per LDS row of x-tile (pad from 96)
#define HSTRIDE 72    // bf16 elements per LDS row of h-tile (pad from 64)

// ---------- prep: init accumulators + convert/swizzle weights to WMMA B-frag order ----------
__global__ void prep_kernel(const float* __restrict__ kW1, const float* __restrict__ kWs,
                            const float* __restrict__ kW2, const float* __restrict__ vW1,
                            const float* __restrict__ vWs, const float* __restrict__ vW2,
                            unsigned short* __restrict__ wbuf,
                            float* __restrict__ out, float* __restrict__ segmax,
                            float* __restrict__ denom) {
    const int gid = blockIdx.x * blockDim.x + threadIdx.x;
    const int stride = gridDim.x * blockDim.x;

    for (int i = gid; i < N_NODES * OUT_DIM; i += stride) out[i] = 0.0f;
    for (int i = gid; i < N_NODES * HEADS; i += stride) {
        segmax[i] = -INFINITY;
        denom[i]  = 0.0f;
    }
    // 96-K matrices: frag layout idx = ((kk*4+n)*32 + lane)*16 + i ; K = kk*32 + (lane/16)*16 + i
    for (int i = gid; i < 6144; i += stride) {
        int i16 = i & 15;
        int L   = (i >> 4) & 31;
        int n   = (i >> 9) & 3;
        int kk  = i >> 11;
        int K   = kk * 32 + ((L >> 4) << 4) + i16;
        int Ncol = n * 16 + (L & 15);
        float a = (K < IN_DIM) ? kW1[K * OUT_DIM + Ncol] : 0.0f;
        float b = (K < IN_DIM) ? kWs[K * OUT_DIM + Ncol] : 0.0f;
        float c = (K < IN_DIM) ? vW1[K * OUT_DIM + Ncol] : 0.0f;
        float d = (K < IN_DIM) ? vWs[K * OUT_DIM + Ncol] : 0.0f;
        wbuf[OFF_K1 + i] = f2bf(a);
        wbuf[OFF_KS + i] = f2bf(b);
        wbuf[OFF_V1 + i] = f2bf(c);
        wbuf[OFF_VS + i] = f2bf(d);
    }
    // 64-K matrices (W2): kk in 0..1
    for (int i = gid; i < 4096; i += stride) {
        int i16 = i & 15;
        int L   = (i >> 4) & 31;
        int n   = (i >> 9) & 3;
        int kk  = i >> 11;
        int K   = kk * 32 + ((L >> 4) << 4) + i16;
        int Ncol = n * 16 + (L & 15);
        wbuf[OFF_K2 + i] = f2bf(kW2[K * OUT_DIM + Ncol]);
        wbuf[OFF_V2 + i] = f2bf(vW2[K * OUT_DIM + Ncol]);
    }
}

// ---------- B fragment load: 2x b128, coalesced (frag-ordered weights) ----------
__device__ __forceinline__ Frag loadB(const unsigned short* __restrict__ wf, int fragIdx, int lane) {
    Frag f;
    const uint4* p = reinterpret_cast<const uint4*>(wf + fragIdx * 512 + lane * 16);
    f.u[0] = p[0];
    f.u[1] = p[1];
    return f;
}

// ---------- one residual block (16 edges x 64 out) via WMMA bf16 ----------
__device__ __forceinline__ Acc4 residual_wmma(const unsigned short* __restrict__ xsh,
                                              unsigned short* __restrict__ hsh,
                                              const unsigned short* __restrict__ w1f,
                                              const unsigned short* __restrict__ wsf,
                                              const unsigned short* __restrict__ w2f,
                                              const float* __restrict__ b1,
                                              int half, int col16, int lane) {
    const v8f vzero = {0.f, 0.f, 0.f, 0.f, 0.f, 0.f, 0.f, 0.f};
    v8f accH[4];
    Acc4 acc;
#pragma unroll
    for (int n = 0; n < 4; ++n) { accH[n] = vzero; acc.a[n] = vzero; }

    // layer 1: h_pre = x@W1 ; skip = x@Ws   (K = 96, 3 blocks of 32)
#pragma unroll
    for (int kk = 0; kk < 3; ++kk) {
        Frag a;
        const int c0 = kk * 32 + half * 16;
        a.u[0] = *reinterpret_cast<const uint4*>(xsh + col16 * XSTRIDE + c0);
        a.u[1] = *reinterpret_cast<const uint4*>(xsh + col16 * XSTRIDE + c0 + 8);
#pragma unroll
        for (int n = 0; n < 4; ++n) {
            Frag b1f = loadB(w1f, kk * 4 + n, lane);
            accH[n] = __builtin_amdgcn_wmma_f32_16x16x32_bf16(
                false, a.v, false, b1f.v, (short)0, accH[n], false, false);
            Frag bsf = loadB(wsf, kk * 4 + n, lane);
            acc.a[n] = __builtin_amdgcn_wmma_f32_16x16x32_bf16(
                false, a.v, false, bsf.v, (short)0, acc.a[n], false, false);
        }
    }

    // h = relu(h_pre + b1) -> LDS (C-layout -> row-major bf16)
#pragma unroll
    for (int n = 0; n < 4; ++n) {
        const float bb = b1[n * 16 + col16];
#pragma unroll
        for (int r = 0; r < 8; ++r) {
            float hv = accH[n][r] + bb;
            hv = hv > 0.0f ? hv : 0.0f;
            hsh[(r + half * 8) * HSTRIDE + n * 16 + col16] = f2bf(hv);
        }
    }
    __syncthreads();

    // layer 2: acc += h@W2   (K = 64, 2 blocks of 32)
#pragma unroll
    for (int kk = 0; kk < 2; ++kk) {
        Frag a;
        const int c0 = kk * 32 + half * 16;
        a.u[0] = *reinterpret_cast<const uint4*>(hsh + col16 * HSTRIDE + c0);
        a.u[1] = *reinterpret_cast<const uint4*>(hsh + col16 * HSTRIDE + c0 + 8);
#pragma unroll
        for (int n = 0; n < 4; ++n) {
            Frag b2f = loadB(w2f, kk * 4 + n, lane);
            acc.a[n] = __builtin_amdgcn_wmma_f32_16x16x32_bf16(
                false, a.v, false, b2f.v, (short)0, acc.a[n], false, false);
        }
    }
    return acc;
}

// ---------- main WMMA edge kernel: k/v residual blocks + scores + segmax + v store ----------
__global__ __launch_bounds__(256) void edge_kernel(
    const float* __restrict__ xs, const float* __restrict__ xd,
    const float* __restrict__ ea, const int* __restrict__ idx,
    const float* __restrict__ qg,
    const float* __restrict__ kb1, const float* __restrict__ kb2,
    const float* __restrict__ vb1, const float* __restrict__ vb2,
    const unsigned short* __restrict__ wbuf,
    float* __restrict__ scores, unsigned short* __restrict__ vvals,
    float* __restrict__ segmax) {

    __shared__ unsigned short xtile[8 * 16 * XSTRIDE];
    __shared__ unsigned short htile[8 * 16 * HSTRIDE];

    const int lane  = threadIdx.x & 31;
    const int wave  = threadIdx.x >> 5;
    const int half  = lane >> 4;
    const int col16 = lane & 15;
    const int ebase = blockIdx.x * 128 + wave * 16;

    unsigned short* xsh = xtile + wave * 16 * XSTRIDE;
    unsigned short* hsh = htile + wave * 16 * HSTRIDE;

    // stage x = [x_src | x_dst | edge_attr | 0-pad] as bf16, 16 rows x 96 cols
    for (int t = lane; t < 16 * KPAD; t += 32) {
        const int r = t / KPAD;
        const int c = t % KPAD;
        const int e = ebase + r;
        float val;
        if (c < 32)      val = xs[e * 32 + c];
        else if (c < 64) val = xd[e * 32 + (c - 32)];
        else if (c < 80) val = ea[e * 16 + (c - 64)];
        else             val = 0.0f;
        xsh[r * XSTRIDE + c] = f2bf(val);
    }
    __syncthreads();

    // ----- K residual block + per-head scores -----
    Acc4 kacc = residual_wmma(xsh, hsh, wbuf + OFF_K1, wbuf + OFF_KS, wbuf + OFF_K2,
                              kb1, half, col16, lane);
#pragma unroll
    for (int n = 0; n < 4; ++n) {  // head == n (DH == 16)
        const float b2v = kb2[n * 16 + col16];
        const float qv  = qg[n * 16 + col16];
        float p[8];
#pragma unroll
        for (int r = 0; r < 8; ++r) p[r] = (kacc.a[n][r] + b2v) * qv;
#pragma unroll
        for (int m = 1; m < 16; m <<= 1) {
#pragma unroll
            for (int r = 0; r < 8; ++r) p[r] += __shfl_xor(p[r], m, 16);
        }
        if (col16 == 0) {
#pragma unroll
            for (int r = 0; r < 8; ++r) {
                const int e = ebase + r + half * 8;
                const float sc = p[r] * 0.25f;  // 1/sqrt(DH)
                scores[e * 4 + n] = sc;
                atomicMaxF(&segmax[(long)idx[e] * 4 + n], sc);
            }
        }
    }
    __syncthreads();

    // ----- V residual block + bf16 store -----
    Acc4 vacc = residual_wmma(xsh, hsh, wbuf + OFF_V1, wbuf + OFF_VS, wbuf + OFF_V2,
                              vb1, half, col16, lane);
#pragma unroll
    for (int n = 0; n < 4; ++n) {
        const float b2v = vb2[n * 16 + col16];
#pragma unroll
        for (int r = 0; r < 8; ++r) {
            const long e = ebase + r + half * 8;
            vvals[e * OUT_DIM + n * 16 + col16] = f2bf(vacc.a[n][r] + b2v);
        }
    }
}

// ---------- exp + denom ----------
__global__ void exp_kernel(const int* __restrict__ idx, float* __restrict__ scores,
                           const float* __restrict__ segmax, float* __restrict__ denom) {
    const int t = blockIdx.x * blockDim.x + threadIdx.x;
    if (t >= E_EDGES * HEADS) return;
    const int e = t >> 2;
    const int h = t & 3;
    const int node = idx[e];
    float m = segmax[node * 4 + h];
    if (!isfinite(m)) m = 0.0f;
    const float ex = __expf(scores[t] - m);
    scores[t] = ex;
    atomicAdd(&denom[node * 4 + h], ex);
}

// ---------- alpha-weighted scatter ----------
__global__ void scatter_kernel(const int* __restrict__ idx, const float* __restrict__ scores,
                               const float* __restrict__ denom,
                               const unsigned short* __restrict__ vvals,
                               float* __restrict__ out) {
    const int e = blockIdx.x * blockDim.x + threadIdx.x;
    if (e >= E_EDGES) return;
    const int node = idx[e];
    float alpha[4];
#pragma unroll
    for (int h = 0; h < 4; ++h)
        alpha[h] = scores[e * 4 + h] / (denom[node * 4 + h] + 1e-16f);
    const unsigned short* vrow = vvals + (long)e * OUT_DIM;
    float* orow = out + (long)node * OUT_DIM;
#pragma unroll
    for (int d = 0; d < OUT_DIM; ++d) {
        atomicAdd(&orow[d], alpha[d >> 4] * bf2f(vrow[d]));
    }
}

extern "C" void kernel_launch(void* const* d_in, const int* in_sizes, int n_in,
                              void* d_out, int out_size, void* d_ws, size_t ws_size,
                              hipStream_t stream) {
    (void)in_sizes; (void)n_in; (void)out_size; (void)ws_size;
    const float* xs  = (const float*)d_in[0];
    const float* xd  = (const float*)d_in[1];
    const float* ea  = (const float*)d_in[2];
    const int*   idx = (const int*)d_in[3];
    const float* q   = (const float*)d_in[4];
    const float* kW1 = (const float*)d_in[5];
    const float* kb1 = (const float*)d_in[6];
    const float* kW2 = (const float*)d_in[7];
    const float* kb2 = (const float*)d_in[8];
    const float* kWs = (const float*)d_in[9];
    const float* vW1 = (const float*)d_in[10];
    const float* vb1 = (const float*)d_in[11];
    const float* vW2 = (const float*)d_in[12];
    const float* vb2 = (const float*)d_in[13];
    const float* vWs = (const float*)d_in[14];
    float* out = (float*)d_out;

    // workspace carve-up
    char* ws = (char*)d_ws;
    unsigned short* wbuf  = (unsigned short*)ws;                       // 64 KB
    float*          sbuf  = (float*)(ws + WBUF_USHORTS * 2);           // E*4 f32
    unsigned short* vvals = (unsigned short*)(ws + WBUF_USHORTS * 2
                              + (size_t)E_EDGES * HEADS * 4);          // E*64 bf16
    float*          segmx = (float*)(ws + WBUF_USHORTS * 2
                              + (size_t)E_EDGES * HEADS * 4
                              + (size_t)E_EDGES * OUT_DIM * 2);        // N*4 f32
    float*          denom = segmx + (size_t)N_NODES * HEADS;           // N*4 f32

    prep_kernel<<<2048, 256, 0, stream>>>(kW1, kWs, kW2, vW1, vWs, vW2,
                                          wbuf, out, segmx, denom);

    edge_kernel<<<E_EDGES / 128, 256, 0, stream>>>(
        xs, xd, ea, idx, q, kb1, kb2, vb1, vb2, wbuf, sbuf, vvals, segmx);

    exp_kernel<<<(E_EDGES * HEADS + 255) / 256, 256, 0, stream>>>(idx, sbuf, segmx, denom);

    scatter_kernel<<<(E_EDGES + 255) / 256, 256, 0, stream>>>(idx, sbuf, denom, vvals, out);
}
